// Upsample_14766097564202
// MI455X (gfx1250) — matrix operand order repaired and checked by reading, not compiled, not run on hardware
//
#include <hip/hip_runtime.h>

// ---------------------------------------------------------------------------
// Fused upsample_conv_2d (StyleGAN2 / score-SDE) for MI455X (gfx1250).
//
// conv_transpose2d(x,w,stride=2) + depthwise FIR outer([1,3,3,1])^2/16 + bias
//   ==  4 polyphase 3x3 convs at coarse resolution, taps {-1,0,+1}:
//   Wp[py,px][dy,dx] = (1/16) sum_{i,j} C[py][dy][i]*C[px][dx][j]*w[i,j,:,:]
//
// Compute-bound (AI ~920 FLOP/B) -> bf16 matrix pipe with the bf16x3
// error-compensated split (fp32-class accuracy, 2.67x fewer matrix ops than
// the f32 16x16x4 pipe):  A*B ~= Ahi*Bhi + Ahi*Blo + Alo*Bhi, fp32 C/D.
// Wave tiling 4 M-tiles x 4 N-tiles: 256 B of operands per WMMA (~32 B/cyc
// per SIMD at 8-cyc WMMA issue), balanced A/B register-level reuse.
// ---------------------------------------------------------------------------

typedef __attribute__((ext_vector_type(4)))  float  v4f;
typedef __attribute__((ext_vector_type(8)))  float  v8f;
typedef __attribute__((ext_vector_type(16))) __bf16 v16bf;

// Polyphase mixing coefficients: CTAB[parity][tap d=-1,0,+1][conv index i]
__device__ __constant__ int CTAB[2][3][3] = {
    {{0, 1, 3}, {3, 3, 1}, {1, 0, 0}},   // fine-coordinate parity 0
    {{0, 0, 1}, {1, 3, 3}, {3, 1, 0}},   // fine-coordinate parity 1
};

#define CIN   256
#define COUT  256
#define Hc    128
#define Wc    128
#define NSEG  36   // 4 phases * 9 taps
// bf16 fragment pool: [g(36)][ntg(16)][kk(8)][lane(32)][e(16)]  (hi and lo)
#define WT_ELEMS ((size_t)NSEG * 16 * 8 * 32 * 16)   // 2,359,296 bf16 each

// ---------------------------------------------------------------------------
// Kernel 1: build phase weights, split fp32 -> bf16 hi/lo, and store in the
// 32x16 bf16 B layout with contiguous k (same order as the A loads):
//   lane L = nIdx + 16*h holds column co = 16*ntg + nIdx,
//   element e (2 per VGPR) holds K-channel ci = 32*kk + 16*h + e.
// One fragment = 32 lanes * 32B -> per-lane b128 pair in the GEMM.
// ---------------------------------------------------------------------------
__global__ __launch_bounds__(256) void phase_weight_pack_bf16(
    const float* __restrict__ w,     // (3,3,256,256) HWIO
    __bf16* __restrict__ wh,
    __bf16* __restrict__ wl)
{
    const int idx = blockIdx.x * 256 + threadIdx.x;   // < 2,359,296
    const int e   = idx & 15;
    const int L   = (idx >> 4) & 31;
    const int kk  = (idx >> 9) & 7;
    const int ntg = (idx >> 12) & 15;
    const int g   = idx >> 16;
    const int h   = L >> 4;
    const int ci  = kk * 32 + h * 16 + e;
    const int co  = ntg * 16 + (L & 15);
    const int pp  = g / 9;
    const int tap = g - pp * 9;
    const int ty  = tap / 3;
    const int tx  = tap - ty * 3;
    const int py  = pp >> 1;
    const int px  = pp & 1;

    float acc = 0.0f;
    #pragma unroll
    for (int i = 0; i < 3; ++i) {
        const int cy = CTAB[py][ty][i];
        #pragma unroll
        for (int j = 0; j < 3; ++j) {
            const int cx = CTAB[px][tx][j];
            acc += (float)(cy * cx) * w[((i * 3 + j) * CIN + ci) * COUT + co];
        }
    }
    acc *= (1.0f / 16.0f);
    const __bf16 hi = (__bf16)acc;
    const __bf16 lo = (__bf16)(acc - (float)hi);
    wh[idx] = hi;
    wl[idx] = lo;
}

// ---------------------------------------------------------------------------
// Kernel 2: main fused conv. Grid (t=128, py=2, n=4), 256 threads (8 waves).
// Wave = (mg, nh): 4 M-tiles (64 coarse pixels) x 4 N-tiles (64 channels),
// px parity looped. Each B fragment feeds 4 M-tiles (12 WMMAs per hi/lo
// pair); each A fragment feeds 4 N-tiles.
// ---------------------------------------------------------------------------
__global__ __launch_bounds__(256) void upfirdn_conv_wmma_bf16(
    const float*  __restrict__ x,     // (4,128,128,256)
    const __bf16* __restrict__ wh,    // packed hi fragments
    const __bf16* __restrict__ wl,    // packed lo fragments
    const float*  __restrict__ bias,  // (256,)
    float*        __restrict__ out)   // (4,256,256,256)
{
    const int t    = blockIdx.x;            // coarse row
    const int py   = blockIdx.y;            // fine-row parity
    const int n    = blockIdx.z;            // batch
    const int wave = threadIdx.x >> 5;
    const int lane = threadIdx.x & 31;
    const int m    = lane & 15;             // M / N index within a tile
    const int lh   = lane >> 4;             // lane half
    const int mg   = wave >> 2;             // 0..1 : 64-pixel group
    const int nh   = wave & 3;              // 0..3 : 64-channel quarter
    const int s0   = mg << 6;               // first coarse pixel
    const int co0  = nh << 6;               // first out channel

    float bv[4];
    #pragma unroll
    for (int nt = 0; nt < 4; ++nt) bv[nt] = bias[co0 + nt * 16 + m];

    const int    Y      = 2 * t + py;
    const size_t xplane = (size_t)n * (Hc * Wc * CIN);

    for (int px = 0; px < 2; ++px) {
        v8f acc[4][4];
        const v8f vzero = {0.f, 0.f, 0.f, 0.f, 0.f, 0.f, 0.f, 0.f};
        #pragma unroll
        for (int a = 0; a < 4; ++a)
            #pragma unroll
            for (int b = 0; b < 4; ++b) acc[a][b] = vzero;

        #pragma unroll 1
        for (int ty = 0; ty < 3; ++ty) {
            const int row = t - 1 + ty;
            if ((unsigned)row >= (unsigned)Hc) continue;   // zero tap (uniform)
            const float* xrow = x + xplane + (size_t)row * (Wc * CIN);

            #pragma unroll 1
            for (int tx = 0; tx < 3; ++tx) {
                const int g = (py * 2 + px) * 9 + ty * 3 + tx;

                // Per-M-tile clamped pixel pointers + zero-pad scales.
                const float* xpix[4];
                float asc[4];
                #pragma unroll
                for (int mt = 0; mt < 4; ++mt) {
                    const int  col  = s0 + mt * 16 + m - 1 + tx;
                    const bool ok   = (unsigned)col < (unsigned)Wc;
                    const int  colc = ok ? col : (col < 0 ? 0 : Wc - 1);
                    xpix[mt] = xrow + ((size_t)colc << 8);
                    asc[mt]  = ok ? 1.0f : 0.0f;
                }

                const size_t segoff =
                    ((size_t)((g * 16 + nh * 4) * 8) << 9) + ((size_t)lane << 4);
                __builtin_prefetch(wh + segoff, 0, 1);

                #pragma unroll 1
                for (int kk = 0; kk < 8; ++kk) {           // K = 256 / 32
                    // ---- A hi/lo fragments (contiguous-k A layout:
                    //      K = 16h+e, matching B pack) ----
                    v16bf ah[4], al[4];
                    #pragma unroll
                    for (int mt = 0; mt < 4; ++mt) {
                        const float* ap = xpix[mt] + (kk << 5) + (lh << 4);
                        v4f f0 = *(const v4f*)(ap);
                        v4f f1 = *(const v4f*)(ap + 4);
                        v4f f2 = *(const v4f*)(ap + 8);
                        v4f f3 = *(const v4f*)(ap + 12);
                        float fv[16];
                        *(v4f*)(&fv[0])  = f0;
                        *(v4f*)(&fv[4])  = f1;
                        *(v4f*)(&fv[8])  = f2;
                        *(v4f*)(&fv[12]) = f3;
                        #pragma unroll
                        for (int e = 0; e < 16; ++e) {
                            const float  v  = fv[e] * asc[mt];
                            const __bf16 hi = (__bf16)v;
                            ah[mt][e] = hi;
                            al[mt][e] = (__bf16)(v - (float)hi);
                        }
                    }

                    const size_t boff = segoff + ((size_t)kk << 9);
                    #pragma unroll
                    for (int nt = 0; nt < 4; ++nt) {
                        const v16bf bh = *(const v16bf*)(wh + boff + ((size_t)nt << 12));
                        const v16bf bl = *(const v16bf*)(wl + boff + ((size_t)nt << 12));
                        #pragma unroll
                        for (int mt = 0; mt < 4; ++mt) {
                            acc[mt][nt] = __builtin_amdgcn_wmma_f32_16x16x32_bf16(
                                false, ah[mt], false, bh, (short)0, acc[mt][nt],
                                false, false);
                            acc[mt][nt] = __builtin_amdgcn_wmma_f32_16x16x32_bf16(
                                false, ah[mt], false, bl, (short)0, acc[mt][nt],
                                false, false);
                            acc[mt][nt] = __builtin_amdgcn_wmma_f32_16x16x32_bf16(
                                false, al[mt], false, bh, (short)0, acc[mt][nt],
                                false, false);
                        }
                    }
                }
            }
        }

        // Store: D tile layout -> lane holds N = m, M = r + 8*lh for VGPR r.
        const size_t orow = ((size_t)(n * 256 + Y)) << 8;
        #pragma unroll
        for (int mt = 0; mt < 4; ++mt) {
            #pragma unroll
            for (int nt = 0; nt < 4; ++nt) {
                #pragma unroll
                for (int r = 0; r < 8; ++r) {
                    const int pix = s0 + mt * 16 + r + (lh << 3);
                    const int X   = 2 * pix + px;
                    out[((orow + X) << 8) + co0 + nt * 16 + m] =
                        acc[mt][nt][r] + bv[nt];
                }
            }
        }
    }
}

// ---------------------------------------------------------------------------
extern "C" void kernel_launch(void* const* d_in, const int* in_sizes, int n_in,
                              void* d_out, int out_size, void* d_ws,
                              size_t ws_size, hipStream_t stream)
{
    const float* x  = (const float*)d_in[0];   // 4*128*128*256
    const float* w  = (const float*)d_in[1];   // 3*3*256*256
    const float* b  = (const float*)d_in[2];   // 256
    float*       o  = (float*)d_out;           // 4*256*256*256

    __bf16* wh = (__bf16*)d_ws;                // 2,359,296 bf16
    __bf16* wl = wh + WT_ELEMS;                // 2,359,296 bf16

    if (ws_size < 2 * WT_ELEMS * sizeof(__bf16)) return;  // deterministic no-op

    // 1) phase-weight transform + bf16 hi/lo fragment packing
    phase_weight_pack_bf16<<<(int)(WT_ELEMS / 256), 256, 0, stream>>>(w, wh, wl);

    // 2) fused polyphase conv, bf16x3 on the 16x16x32 WMMA pipe
    dim3 grid(Hc, 2, 4);
    upfirdn_conv_wmma_bf16<<<grid, 256, 0, stream>>>(x, wh, wl, b, o);
}
